// SwinTransformer1_31258771980373
// MI455X (gfx1250) — matrix operand order repaired and checked by reading, compiled, not run on hardware
//
#include <hip/hip_runtime.h>

typedef __attribute__((ext_vector_type(16))) __bf16 v16bf;
typedef __attribute__((ext_vector_type(8)))  float  v8f;

union Frag {
  v16bf h;
  uint4 u[2];
};

// Native f32 -> bf16 (RNE) via hardware cvt instead of manual bit rounding.
__device__ __forceinline__ unsigned short f2bf(float f) {
  __bf16 h = (__bf16)f;
  unsigned short r;
  __builtin_memcpy(&r, &h, 2);
  return r;
}

// A-matrix 16x32 bf16 fragment from row-major bf16 (lane half K-interleave per ISA 7.12.2)
__device__ __forceinline__ Frag load_a(const unsigned short* base, int stride,
                                       int r0, int k0, int lane) {
  int m = lane & 15, half = (lane >> 4) & 1;
  const unsigned short* p = base + (r0 + m) * stride + k0 + half * 8;
  Frag f;
  f.u[0] = *(const uint4*)(p);        // K = k0 + half*8 + [0..7]
  f.u[1] = *(const uint4*)(p + 16);   // K = k0 + 16 + half*8 + [0..7]
  return f;
}

// B-matrix 32x16 bf16 fragment: B[k][n] = src[n0+n][k0+k], row-major src
__device__ __forceinline__ Frag load_b(const unsigned short* base, int stride,
                                       int n0, int k0, int lane) {
  int n = lane & 15, half = (lane >> 4) & 1;
  const unsigned short* p = base + (n0 + n) * stride + k0 + half * 8;
  Frag f;
  f.u[0] = *(const uint4*)(p);        // K = k0 + half*8 + [0..7]
  f.u[1] = *(const uint4*)(p + 16);   // K = k0 + 16 + half*8 + [0..7]
  return f;
}

__device__ __forceinline__ v8f wmma_bf16(const Frag& a, const Frag& b, v8f c) {
  return __builtin_amdgcn_wmma_f32_16x16x32_bf16(false, a.h, false, b.h,
                                                 (short)0, c, false, false);
}

// ---------------------------------------------------------------------------
// Prep kernel 1: comb[w][h][n][m] = rpb[rel_idx[n,m], h] + mask[w, n, m]
// padded to 64x64; m >= 49 -> -1e30 (softmax kill), n clamped (rows unused)
// ---------------------------------------------------------------------------
__global__ __launch_bounds__(256) void prep_comb(const float* __restrict__ rpb,
                                                 const int* __restrict__ rel_idx,
                                                 const float* __restrict__ mask,
                                                 float* __restrict__ comb) {
  int i = blockIdx.x * 256 + threadIdx.x;          // 64*4*64*64 = 1048576
  if (i >= 64 * 4 * 64 * 64) return;
  int m = i & 63;
  int n = (i >> 6) & 63;
  int h = (i >> 12) & 3;
  int w = i >> 14;
  float v;
  if (m < 49) {
    int nc = n < 49 ? n : 48;
    int idx = rel_idx[nc * 49 + m];
    v = rpb[idx * 4 + h] + mask[(w * 49 + nc) * 49 + m];
  } else {
    v = -1e30f;
  }
  comb[i] = v;
}

// Prep kernel 2: weights f32 -> bf16 (qkv_w [384*128], proj_w [128*128])
__global__ __launch_bounds__(256) void prep_w(const float* __restrict__ qkvw,
                                              const float* __restrict__ projw,
                                              unsigned short* __restrict__ qkvw_bf,
                                              unsigned short* __restrict__ projw_bf) {
  int i = blockIdx.x * 256 + threadIdx.x;
  if (i < 384 * 128) qkvw_bf[i] = f2bf(qkvw[i]);
  if (i < 128 * 128) projw_bf[i] = f2bf(projw[i]);
}

// ---------------------------------------------------------------------------
// Main kernel: one block per window, 8 wave32 per block.
// ---------------------------------------------------------------------------
#define XS_STR 136   // 64x136 bf16, 272B row (16B aligned)
#define VT_STR 72    // 128x72 bf16 (v transposed: [d][token]), 144B row
#define PS_STR 72    // per-head P: [64][72]

__global__ __launch_bounds__(256) void swin_attn(
    const float* __restrict__ x,
    const float* __restrict__ qkv_b,
    const float* __restrict__ proj_b,
    const float* __restrict__ comb,
    const unsigned short* __restrict__ qkvw,
    const unsigned short* __restrict__ projw,
    float* __restrict__ out) {
  __shared__ unsigned short xs[64 * XS_STR];
  __shared__ unsigned short qs[64 * XS_STR];
  __shared__ unsigned short ks[64 * XS_STR];
  __shared__ unsigned short vt[128 * VT_STR];
  __shared__ unsigned short ps[4 * 64 * PS_STR];
  __shared__ unsigned short os[64 * XS_STR];

  const int tid  = threadIdx.x;
  const int lane = tid & 31;
  const int wv   = tid >> 5;
  const int b    = blockIdx.x;
  const int widx = b & 63;               // window-in-image index (nW = 64)
  const int col  = lane & 15;
  const int half = (lane >> 4) & 1;

  // ---- Phase 0: stage x [49][128] f32 -> bf16 LDS, zero rows 49..63 ----
  {
    const float* xb = x + (size_t)b * (49 * 128);
    for (int i = tid; i < 1568; i += 256) {       // 1568 float4 = 6272 floats
      int e = i * 4;
      int r = e >> 7, c = e & 127;
      float4 v = *(const float4*)(xb + e);
      unsigned u0 = (unsigned)f2bf(v.x) | ((unsigned)f2bf(v.y) << 16);
      unsigned u1 = (unsigned)f2bf(v.z) | ((unsigned)f2bf(v.w) << 16);
      *(uint2*)(xs + r * XS_STR + c) = make_uint2(u0, u1);
    }
    for (int i = tid; i < 15 * XS_STR; i += 256) {
      xs[(49 + i / XS_STR) * XS_STR + (i % XS_STR)] = 0;
    }
  }
  __syncthreads();

  // ---- Phase 1: qkv = x @ qkv_w^T + b  (96 tiles of 16x16, K=128) ----
  const float scale = 0.17677669529663687f;       // hd^-0.5
  for (int j = 0; j < 12; ++j) {
    int idx = wv * 12 + j;
    int rt = idx / 24, ct = idx % 24;             // 4 row tiles x 24 col tiles
    v8f acc = {};
    for (int kk = 0; kk < 4; ++kk) {
      Frag A = load_a(xs, XS_STR, rt * 16, kk * 32, lane);
      Frag B = load_b(qkvw, 128, ct * 16, kk * 32, lane);
      acc = wmma_bf16(A, B, acc);
    }
    float bias = qkv_b[ct * 16 + col];
    int r0 = rt * 16 + half * 8;
    if (ct < 8) {                                  // q (pre-scaled)
      for (int v = 0; v < 8; ++v)
        qs[(r0 + v) * XS_STR + ct * 16 + col] = f2bf((acc[v] + bias) * scale);
    } else if (ct < 16) {                          // k row-major [token][d]
      for (int v = 0; v < 8; ++v)
        ks[(r0 + v) * XS_STR + (ct - 8) * 16 + col] = f2bf(acc[v] + bias);
    } else {                                       // v transposed [d][token]
      int c = (ct - 16) * 16 + col;
      unsigned short t[8] __attribute__((aligned(16)));
      for (int v = 0; v < 8; ++v) t[v] = f2bf(acc[v] + bias);
      *(uint4*)(vt + c * VT_STR + r0) = *(const uint4*)t;
    }
  }
  __syncthreads();

  // ---- Phase 2: scores = q @ k^T + comb; softmax; P -> bf16 LDS ----
  {
    int h = wv >> 1;
    const float* cb = comb + ((size_t)(widx * 4 + h)) * 4096;   // [64][64]
    for (int rr = 0; rr < 2; ++rr) {
      int rt = (wv & 1) * 2 + rr;
      Frag A = load_a(qs, XS_STR, rt * 16, h * 32, lane);       // K = hd = 32
      v8f acc[4];
      for (int ctm = 0; ctm < 4; ++ctm) {
        Frag B = load_b(ks, XS_STR, ctm * 16, h * 32, lane);
        v8f z = {};
        acc[ctm] = wmma_bf16(A, B, z);
      }
      float p0[8], p1[8], p2[8], p3[8];
      for (int v = 0; v < 8; ++v) {
        int r = rt * 16 + half * 8 + v;
        float s0 = acc[0][v] + cb[r * 64 + col];
        float s1 = acc[1][v] + cb[r * 64 + 16 + col];
        float s2 = acc[2][v] + cb[r * 64 + 32 + col];
        float s3 = acc[3][v] + cb[r * 64 + 48 + col];
        float mv = fmaxf(fmaxf(s0, s1), fmaxf(s2, s3));
        for (int off = 1; off < 16; off <<= 1)
          mv = fmaxf(mv, __shfl_xor(mv, off));    // reduce within 16-lane half
        float e0 = __expf(s0 - mv), e1 = __expf(s1 - mv);
        float e2 = __expf(s2 - mv), e3 = __expf(s3 - mv);
        float sv = (e0 + e1) + (e2 + e3);
        for (int off = 1; off < 16; off <<= 1)
          sv += __shfl_xor(sv, off);
        float inv = 1.0f / sv;
        p0[v] = e0 * inv; p1[v] = e1 * inv; p2[v] = e2 * inv; p3[v] = e3 * inv;
      }
      unsigned short* pp = ps + h * 64 * PS_STR;
      for (int v = 0; v < 8; ++v) {
        int r = rt * 16 + half * 8 + v;
        pp[r * PS_STR +  0 + col] = f2bf(p0[v]);
        pp[r * PS_STR + 16 + col] = f2bf(p1[v]);
        pp[r * PS_STR + 32 + col] = f2bf(p2[v]);
        pp[r * PS_STR + 48 + col] = f2bf(p3[v]);
      }
    }
  }
  __syncthreads();

  // ---- Phase 3: out_h = P @ v  (32 tiles, K = 64) ----
  for (int j = 0; j < 4; ++j) {
    int idx = wv * 4 + j;                          // 0..31
    int rt = idx >> 3, h = (idx & 7) >> 1, td = idx & 1;
    v8f acc = {};
    for (int kk = 0; kk < 2; ++kk) {
      Frag A = load_a(ps + h * 64 * PS_STR, PS_STR, rt * 16, kk * 32, lane);
      Frag B = load_b(vt, VT_STR, h * 32 + td * 16, kk * 32, lane);
      acc = wmma_bf16(A, B, acc);
    }
    int r0 = rt * 16 + half * 8;
    for (int v = 0; v < 8; ++v)
      os[(r0 + v) * XS_STR + h * 32 + td * 16 + col] = f2bf(acc[v]);
  }
  __syncthreads();

  // ---- Phase 4: final = os @ proj_w^T + proj_b; store f32 rows < 49 ----
  for (int j = 0; j < 4; ++j) {
    int idx = wv * 4 + j;
    int rt = idx >> 3, ct = idx & 7;
    v8f acc = {};
    for (int kk = 0; kk < 4; ++kk) {
      Frag A = load_a(os, XS_STR, rt * 16, kk * 32, lane);
      Frag B = load_b(projw, 128, ct * 16, kk * 32, lane);
      acc = wmma_bf16(A, B, acc);
    }
    float bias = proj_b[ct * 16 + col];
    for (int v = 0; v < 8; ++v) {
      int r = rt * 16 + half * 8 + v;
      if (r < 49)
        out[((size_t)b * 49 + r) * 128 + ct * 16 + col] = acc[v] + bias;
    }
  }
}

// ---------------------------------------------------------------------------
extern "C" void kernel_launch(void* const* d_in, const int* in_sizes, int n_in,
                              void* d_out, int out_size, void* d_ws, size_t ws_size,
                              hipStream_t stream) {
  (void)in_sizes; (void)n_in; (void)out_size; (void)ws_size;
  const float* x      = (const float*)d_in[0];   // [4096,49,128]
  const float* mask   = (const float*)d_in[1];   // [64,49,49]
  const float* qkv_w  = (const float*)d_in[2];   // [384,128]
  const float* qkv_b  = (const float*)d_in[3];   // [384]
  const float* proj_w = (const float*)d_in[4];   // [128,128]
  const float* proj_b = (const float*)d_in[5];   // [128]
  const float* rpb    = (const float*)d_in[6];   // [169,4]
  const int*   rel    = (const int*)d_in[7];     // [49,49]
  float* out = (float*)d_out;

  char* ws = (char*)d_ws;
  float* comb = (float*)ws;                                  // 64*4*64*64 f32 = 4 MB
  unsigned short* qkvw_bf = (unsigned short*)(ws + (size_t)4 * 1024 * 1024);
  unsigned short* projw_bf = qkvw_bf + 384 * 128;

  prep_comb<<<4096, 256, 0, stream>>>(rpb, rel, mask, comb);
  prep_w<<<192, 256, 0, stream>>>(qkv_w, proj_w, qkvw_bf, projw_bf);
  swin_attn<<<4096, 256, 0, stream>>>(x, qkv_b, proj_b, comb, qkvw_bf, projw_bf, out);
}
